// MambaBlock_12764642803900
// MI455X (gfx1250) — matrix operand order repaired and checked by reading, not compiled, not run on hardware
//
#include <hip/hip_runtime.h>

// ---------------- problem constants ----------------
#define B_   2
#define L_   2048
#define DM   512          // d_model
#define DI   1024         // d_inner
#define DS   16           // d_state
#define DTR  32           // dt_rank
#define BL   (B_ * L_)    // 4096 rows
#define CHUNK 64          // scan timesteps staged per TDM tile (64*64*4 = 16KB)

typedef __attribute__((ext_vector_type(16))) _Float16 v16h;
typedef __attribute__((ext_vector_type(8)))  _Float16 v8h;
typedef __attribute__((ext_vector_type(8)))  float    v8f;
typedef unsigned int u32x4 __attribute__((ext_vector_type(4)));
typedef int          i32x4 __attribute__((ext_vector_type(4)));
typedef int          i32x8 __attribute__((ext_vector_type(8)));

#if __has_builtin(__builtin_amdgcn_tensor_load_to_lds)
#define HAVE_TDM 1
#else
#define HAVE_TDM 0
#endif

// ---------------- WMMA fragment helpers (wave32, ISA 7.12.2 layouts) -------

// A-matrix 16x32 f16, row-major source, leading dim lda (halfs).
// lane L: m = L&15, base = (L>>4)*8; v16h[j] = A[m][ j<8 ? base+j : 16+base+j-8 ]
static __device__ inline v16h load_a_frag(const _Float16* A, int row0, int k0, int lda) {
    int lane = threadIdx.x & 31;
    int m    = lane & 15;
    int base = (lane >> 4) * 8;
    const _Float16* p = A + (size_t)(row0 + m) * lda + k0 + base;
    v8h lo = *(const v8h*)(p);        // K = base .. base+7
    v8h hi = *(const v8h*)(p + 16);   // K = 16+base .. 16+base+7
    v16h r;
#pragma unroll
    for (int j = 0; j < 8; ++j) { r[j] = lo[j]; r[j + 8] = hi[j]; }
    return r;
}

// B-matrix 32x16 f16 where Bmat[k][n] = W[n][k], W row-major (N x K), ld=ldw.
// lane L: n = L&15, kb = (L>>4)*8; v16h[j] = W[n][ j<8 ? kb+j : 16+kb+j-8 ]
static __device__ inline v16h load_b_frag(const _Float16* W, int col0, int k0, int ldw) {
    int lane = threadIdx.x & 31;
    int n    = lane & 15;
    int kb   = (lane >> 4) * 8;
    const _Float16* p = W + (size_t)(col0 + n) * ldw + k0 + kb;
    v8h lo = *(const v8h*)(p);
    v8h hi = *(const v8h*)(p + 16);
    v16h r;
#pragma unroll
    for (int j = 0; j < 8; ++j) { r[j] = lo[j]; r[j + 8] = hi[j]; }
    return r;
}

// C/D 16x16 f32: VGPR r holds M = r + 8*(lane>>4), N = lane&15
static __device__ inline void store_c_frag(float* C, int row0, int col0, int ldc, v8f acc) {
    int lane  = threadIdx.x & 31;
    int n     = lane & 15;
    int mbase = (lane >> 4) * 8;
#pragma unroll
    for (int r = 0; r < 8; ++r)
        C[(size_t)(row0 + mbase + r) * ldc + col0 + n] = acc[r];
}

// ---------------- GEMM: C[M,N] = A16[M,K] * W16[N,K]^T (fp32 accum) --------
// 256 threads = 8 waves/block; each wave owns a 16 x (16*NT) strip.
// Software-pipelined: fragments for k+1 are loaded before the WMMAs for k,
// so v_wmma issues with next-step loads still in flight (partial loadcnt wait).
template <int NT>
__global__ void gemm_wmma(const _Float16* __restrict__ A,
                          const _Float16* __restrict__ W,
                          float* __restrict__ C, int M, int N, int K) {
    int wave   = blockIdx.x * 8 + (threadIdx.x >> 5);
    int tilesN = N / (16 * NT);
    int tm     = wave / tilesN;
    int tn     = wave % tilesN;
    if (tm * 16 >= M) return;                 // wave-uniform exit: EXEC all-ones for WMMA
    int col0 = tn * NT * 16;

    v8f acc[NT];
#pragma unroll
    for (int t = 0; t < NT; ++t) acc[t] = (v8f)(0.0f);

    v16h a_cur = load_a_frag(A, tm * 16, 0, K);
    v16h b_cur[NT];
#pragma unroll
    for (int t = 0; t < NT; ++t) b_cur[t] = load_b_frag(W, col0 + t * 16, 0, K);

    int nk = K >> 5;
    for (int kk = 0; kk + 1 < nk; ++kk) {
        int k1 = (kk + 1) << 5;
        v16h a_nxt = load_a_frag(A, tm * 16, k1, K);
        v16h b_nxt[NT];
#pragma unroll
        for (int t = 0; t < NT; ++t) b_nxt[t] = load_b_frag(W, col0 + t * 16, k1, K);
#pragma unroll
        for (int t = 0; t < NT; ++t)
            acc[t] = __builtin_amdgcn_wmma_f32_16x16x32_f16(
                false, a_cur, false, b_cur[t], (short)0, acc[t], false, false);
        a_cur = a_nxt;
#pragma unroll
        for (int t = 0; t < NT; ++t) b_cur[t] = b_nxt[t];
    }
#pragma unroll
    for (int t = 0; t < NT; ++t)
        acc[t] = __builtin_amdgcn_wmma_f32_16x16x32_f16(
            false, a_cur, false, b_cur[t], (short)0, acc[t], false, false);

#pragma unroll
    for (int t = 0; t < NT; ++t)
        store_c_frag(C, tm * 16, col0 + t * 16, N, acc[t]);
}

// ---------------- misc elementwise kernels ----------------
__global__ void f32_to_f16_kernel(const float* __restrict__ src,
                                  _Float16* __restrict__ dst, int n) {
    int i = blockIdx.x * blockDim.x + threadIdx.x;
    if (i < n) dst[i] = (_Float16)src[i];
}

// LayerNorm over D=512, one block (256 thr) per row; also copies residual.
__global__ void layernorm_kernel(const float* __restrict__ x,
                                 const float* __restrict__ w,
                                 const float* __restrict__ b,
                                 _Float16* __restrict__ h16,
                                 float* __restrict__ resid) {
    int row = blockIdx.x;
    int tid = threadIdx.x;
    const float* xr = x + (size_t)row * DM;
    float v0 = xr[tid], v1 = xr[tid + 256];
    resid[(size_t)row * DM + tid]       = v0;
    resid[(size_t)row * DM + tid + 256] = v1;
    __shared__ float s1[256], s2[256];
    s1[tid] = v0 + v1;
    s2[tid] = v0 * v0 + v1 * v1;
    __syncthreads();
    for (int off = 128; off > 0; off >>= 1) {
        if (tid < off) { s1[tid] += s1[tid + off]; s2[tid] += s2[tid + off]; }
        __syncthreads();
    }
    float mu  = s1[0] * (1.0f / DM);
    float var = s2[0] * (1.0f / DM) - mu * mu;
    float rs  = rsqrtf(var + 1e-5f);
    h16[(size_t)row * DM + tid]       = (_Float16)((v0 - mu) * rs * w[tid] + b[tid]);
    h16[(size_t)row * DM + tid + 256] = (_Float16)((v1 - mu) * rs * w[tid + 256] + b[tid + 256]);
}

// Depthwise causal conv (K=4) + SiLU, reading x = xz[:, 0:DI].
// dir=0: acc += w[k]*x[t-3+k] ; dir=1 (time-reversed branch in original idx): x[t+3-k]
__global__ void conv_silu_kernel(const float* __restrict__ xz,
                                 const float* __restrict__ w,
                                 const float* __restrict__ bias,
                                 _Float16* __restrict__ out16, int dir) {
    int idx = blockIdx.x * blockDim.x + threadIdx.x;  // over B*L*DI
    if (idx >= B_ * L_ * DI) return;
    int d  = idx & (DI - 1);
    int bl = idx >> 10;
    int t  = bl & (L_ - 1);
    int bb = bl >> 11;
    float acc = bias[d];
#pragma unroll
    for (int k = 0; k < 4; ++k) {
        int tt = dir ? (t + 3 - k) : (t - 3 + k);
        if (tt >= 0 && tt < L_)
            acc = fmaf(w[d * 4 + k], xz[((size_t)(bb * L_ + tt)) * (2 * DI) + d], acc);
    }
    acc = acc / (1.0f + __expf(-acc));        // SiLU
    out16[idx] = (_Float16)acc;
}

// ---------------- TDM helper: DMA `rows` x 64 f32 rows into LDS ------------
#if HAVE_TDM
static __device__ inline void tdm_load_rows(unsigned lds_off, const float* gsrc, int rows) {
    unsigned long long ga = (unsigned long long)(uintptr_t)gsrc;
    // D# group 0: count=1, lds_addr, 57-bit global addr, type=2 ("image")
    u32x4 g0;
    g0[0] = 1u;
    g0[1] = lds_off;
    g0[2] = (unsigned)(ga & 0xFFFFFFFFu);
    g0[3] = ((unsigned)(ga >> 32) & 0x01FFFFFFu) | (2u << 30);
    // D# group 1: data_size=4B, tensor 64 x rows (stride 64), tile 64 x rows,
    // no multicast / padding / iterate / atomic-barrier.
    i32x8 g1;
    g1[0] = (2 << 16);      // data_size = 2 -> 4 bytes
    g1[1] = (64 << 16);     // tensor_dim0[15:0] = 64
    g1[2] = (rows << 16);   // tensor_dim1[15:0] = rows
    g1[3] = (64 << 16);     // tile_dim0 = 64
    g1[4] = rows;           // tile_dim1 = rows
    g1[5] = 64;             // tensor_dim0_stride = 64
    g1[6] = 0;
    g1[7] = 0;
    i32x4 z4 = {0, 0, 0, 0};
#if __clang_major__ >= 23
    i32x8 z8 = {0, 0, 0, 0, 0, 0, 0, 0};
    __builtin_amdgcn_tensor_load_to_lds(g0, g1, z4, z4, z8, 0);
#else
    __builtin_amdgcn_tensor_load_to_lds(g0, g1, z4, z4, 0);
#endif
}
#endif

// Fused selective scan: dt_proj+softplus, dA=exp(dt*A), state recurrence,
// y = sum_n h*C + x*D, gated by SiLU(z). One lane per channel, 16 states +
// 32 dt weights in registers. x_dbl staged in CHUNK-step LDS tiles via the
// Tensor Data Mover (double-buffered; wave 0 issues DMA for chunk c+1 while
// all waves consume chunk c; synced with s_wait_tensorcnt + barrier).
__global__ void ssm_scan_kernel(const _Float16* __restrict__ xc16,
                                const float* __restrict__ xdbl,
                                const float* __restrict__ xz,   // z = cols DI..2DI
                                const float* __restrict__ Wdt,  // (DI, DTR)
                                const float* __restrict__ dt_bias,
                                const float* __restrict__ A_log,
                                const float* __restrict__ Dskip,
                                float* __restrict__ yout, int dir) {
    __shared__ __align__(16) float sbuf[2][CHUNK * 64];
    int bb = blockIdx.y;
    int d  = blockIdx.x * 256 + threadIdx.x;   // 0..DI-1
    bool issuer = (threadIdx.x >> 5) == 0;     // wave 0 drives the TDM

    float wdt[DTR];
#pragma unroll
    for (int r = 0; r < DTR; r += 4)
        *(float4*)&wdt[r] = *(const float4*)&Wdt[(size_t)d * DTR + r];

    float Av[DS];
#pragma unroll
    for (int n = 0; n < DS; ++n) Av[n] = -__expf(A_log[(size_t)d * DS + n]);

    float bias = dt_bias[d];
    float Dk   = Dskip[d];
    float h[DS];
#pragma unroll
    for (int n = 0; n < DS; ++n) h[n] = 0.0f;

    const int nch = L_ / CHUNK;
    const size_t rowbase = (size_t)bb * L_;

#if HAVE_TDM
    if (issuer) {
        int lo0 = dir ? (L_ - CHUNK) : 0;
        tdm_load_rows((unsigned)(uintptr_t)&sbuf[0][0],
                      xdbl + (rowbase + lo0) * 64, CHUNK);
        __builtin_amdgcn_s_wait_tensorcnt((short)0);
    }
    __syncthreads();
#endif

    for (int ci = 0; ci < nch; ++ci) {
        int lo = dir ? (L_ - (ci + 1) * CHUNK) : ci * CHUNK;
#if HAVE_TDM
        if (issuer && ci + 1 < nch) {
            int lo1 = dir ? (L_ - (ci + 2) * CHUNK) : (ci + 1) * CHUNK;
            tdm_load_rows((unsigned)(uintptr_t)&sbuf[(ci + 1) & 1][0],
                          xdbl + (rowbase + lo1) * 64, CHUNK);
        }
        const float* sx = &sbuf[ci & 1][0];
#else
        for (int idx = threadIdx.x; idx < CHUNK * 64; idx += 256)
            sbuf[0][idx] = xdbl[(rowbase + lo) * 64 + idx];
        __syncthreads();
        const float* sx = &sbuf[0][0];
#endif
        for (int j = 0; j < CHUNK; ++j) {
            int t = dir ? (lo + CHUNK - 1 - j) : (lo + j);
            const float* row = sx + (t - lo) * 64;
            size_t base = rowbase + t;

            float dtr = bias;
#pragma unroll
            for (int r = 0; r < DTR; ++r) dtr = fmaf(row[r], wdt[r], dtr);
            float dt = (dtr > 20.0f) ? dtr : log1pf(__expf(dtr));   // softplus

            float xv  = (float)xc16[base * DI + d];
            float dtx = dt * xv;
            float y   = 0.0f;
#pragma unroll
            for (int n = 0; n < DS; ++n) {
                float dA = __expf(dt * Av[n]);
                h[n] = fmaf(dA, h[n], dtx * row[DTR + n]);
                y    = fmaf(h[n], row[DTR + DS + n], y);
            }
            y = fmaf(xv, Dk, y);
            float zv = xz[base * (2 * DI) + DI + d];
            y *= zv / (1.0f + __expf(-zv));        // SiLU(z) gate
            yout[base * DI + d] = y;
        }
        __syncthreads();                           // all waves done with sbuf[ci&1]
#if HAVE_TDM
        if (issuer && ci + 1 < nch)
            __builtin_amdgcn_s_wait_tensorcnt((short)0);   // chunk ci+1 landed
        __syncthreads();                           // publish next buffer
#endif
    }
}

__global__ void combine_kernel(const float* __restrict__ yf,
                               const float* __restrict__ yb,
                               _Float16* __restrict__ ysum16, int n) {
    int i = blockIdx.x * blockDim.x + threadIdx.x;
    if (i < n) ysum16[i] = (_Float16)(0.5f * (yf[i] + yb[i]));
}

// ---------------- host orchestration ----------------
extern "C" void kernel_launch(void* const* d_in, const int* in_sizes, int n_in,
                              void* d_out, int out_size, void* d_ws, size_t ws_size,
                              hipStream_t stream) {
    (void)in_sizes; (void)n_in; (void)out_size; (void)ws_size;

    const float* hidden    = (const float*)d_in[0];
    const float* norm_w    = (const float*)d_in[1];
    const float* norm_b    = (const float*)d_in[2];
    const float* in_proj_w = (const float*)d_in[3];   // (2048, 512)
    const float* out_proj_w= (const float*)d_in[4];   // (512, 1024)
    const float* conv_w[2] = {(const float*)d_in[5],  (const float*)d_in[12]};
    const float* conv_b[2] = {(const float*)d_in[6],  (const float*)d_in[13]};
    const float* xp_w[2]   = {(const float*)d_in[7],  (const float*)d_in[14]};
    const float* dt_w[2]   = {(const float*)d_in[8],  (const float*)d_in[15]};
    const float* dt_b[2]   = {(const float*)d_in[9],  (const float*)d_in[16]};
    const float* A_log[2]  = {(const float*)d_in[10], (const float*)d_in[17]};
    const float* Dsk[2]    = {(const float*)d_in[11], (const float*)d_in[18]};

    char* ws = (char*)d_ws;
    size_t off = 0;
    auto alloc = [&](size_t bytes) -> void* {
        void* p = (void*)(ws + off);
        off += (bytes + 255) & ~(size_t)255;
        return p;
    };

    _Float16* Win16    = (_Float16*)alloc((size_t)(2 * DI) * DM * 2);   // 2 MB
    _Float16* Wout16   = (_Float16*)alloc((size_t)DM * DI * 2);         // 1 MB
    _Float16* Wxp16[2] = {(_Float16*)alloc((size_t)64 * DI * 2),
                          (_Float16*)alloc((size_t)64 * DI * 2)};
    _Float16* h16      = (_Float16*)alloc((size_t)BL * DM * 2);         // 4 MB
    float*    xzbuf    = (float*)alloc((size_t)BL * 2 * DI * 4);        // 33.6 MB
    _Float16* xc16[2]  = {(_Float16*)alloc((size_t)BL * DI * 2),
                          (_Float16*)alloc((size_t)BL * DI * 2)};
    float*    xdbl[2]  = {(float*)alloc((size_t)BL * 64 * 4),
                          (float*)alloc((size_t)BL * 64 * 4)};
    float*    ybuf[2]  = {(float*)alloc((size_t)BL * DI * 4),
                          (float*)alloc((size_t)BL * DI * 4)};
    _Float16* ysum16   = (_Float16*)alloc((size_t)BL * DI * 2);         // total ~102 MB

    float* out_f  = (float*)d_out;                 // (B,L,512)
    float* resid  = out_f + (size_t)BL * DM;       // (B,L,512)

    // 1) weight converts to f16
    int nWin = 2 * DI * DM, nWout = DM * DI, nWxp = 64 * DI;
    f32_to_f16_kernel<<<(nWin  + 255) / 256, 256, 0, stream>>>(in_proj_w,  Win16,    nWin);
    f32_to_f16_kernel<<<(nWout + 255) / 256, 256, 0, stream>>>(out_proj_w, Wout16,   nWout);
    f32_to_f16_kernel<<<(nWxp  + 255) / 256, 256, 0, stream>>>(xp_w[0],    Wxp16[0], nWxp);
    f32_to_f16_kernel<<<(nWxp  + 255) / 256, 256, 0, stream>>>(xp_w[1],    Wxp16[1], nWxp);

    // 2) LayerNorm + residual copy
    layernorm_kernel<<<BL, 256, 0, stream>>>(hidden, norm_w, norm_b, h16, resid);

    // 3) in_proj: (4096x512) x (512x2048) -> xz   (8192 wave-tiles -> 1024 blocks)
    gemm_wmma<4><<<1024, 256, 0, stream>>>(h16, Win16, xzbuf, BL, 2 * DI, DM);

    int nEl = B_ * L_ * DI;
    for (int br = 0; br < 2; ++br) {
        // 4) depthwise conv + SiLU -> f16
        conv_silu_kernel<<<(nEl + 255) / 256, 256, 0, stream>>>(
            xzbuf, conv_w[br], conv_b[br], xc16[br], br);
        // 5) x_proj: (4096x1024) x (1024x64) -> x_dbl   (256 wave-tiles -> 32 blocks)
        gemm_wmma<4><<<32, 256, 0, stream>>>(xc16[br], Wxp16[br], xdbl[br], BL, 64, DI);
        // 6) fused dt_proj + selective scan + SiLU(z) gate (TDM-staged x_dbl)
        ssm_scan_kernel<<<dim3(DI / 256, B_), 256, 0, stream>>>(
            xc16[br], xdbl[br], xzbuf, dt_w[br], dt_b[br], A_log[br], Dsk[br],
            ybuf[br], br);
    }

    // 7) combine branches -> f16
    combine_kernel<<<(nEl + 255) / 256, 256, 0, stream>>>(ybuf[0], ybuf[1], ysum16, nEl);

    // 8) out_proj: (4096x1024) x (1024x512) -> out   (2048 wave-tiles -> 256 blocks)
    gemm_wmma<4><<<256, 256, 0, stream>>>(ysum16, Wout16, out_f, BL, DM, DI);
}
